// StarClassifierFusionMambaTokenized_44367012167903
// MI455X (gfx1250) — compile-verified, matrix operand order checked by reading, not run on hardware
//
#include <hip/hip_runtime.h>
#include <hip/hip_bf16.h>
#include <math.h>

// ---------------- model constants ----------------
#define DMODEL   512
#define DSTATE   128
#define DCONV    4
#define NLAYERS  10
#define HEADDIM  64
#define DINNER   1024            // EXPAND * DMODEL
#define NHEADS   16              // DINNER / HEADDIM
#define CONVDIM  1280            // DINNER + 2*DSTATE
#define DPROJ    2320            // 2*DINNER + 2*DSTATE + NHEADS
#define BATCH    8
#define IN_S     3647
#define TOK_S    64
#define NTOK_S   57              // ceil(3647/64)
#define IN_G     18
#define TOK_G    2
#define NTOK_G   9
#define NCLS     55
#define CAHEADS  8
#define EPSF     1e-5f

typedef __attribute__((ext_vector_type(16))) _Float16 v16h;
typedef __attribute__((ext_vector_type(8)))  float    v8f;

// =====================================================================
// Fragment loaders for V_WMMA_F32_16X16X32_F16 (CDNA5 ISA 7.12.2):
//   lanes 0..15  hold K offsets {0..7, 16..23}   (hs = 0)
//   lanes 16..31 hold K offsets {8..15, 24..31}  (hs = 1)
// Row-major contiguous K: a lane's 16 elements are two 8-float runs at
// +0 and +16 from (k0 + hs*8).
// =====================================================================

// branch-free vectorized loader (requires 16B-aligned p, full tile in range)
__device__ __forceinline__ v16h load_frag_contig(const float* __restrict__ p)
{
    const float4* q = (const float4*)p;
    float4 x0 = q[0];   // k +0..3
    float4 x1 = q[1];   // k +4..7
    float4 x2 = q[4];   // k +16..19
    float4 x3 = q[5];   // k +20..23
    v16h f;
    f[0]  = (_Float16)x0.x; f[1]  = (_Float16)x0.y; f[2]  = (_Float16)x0.z; f[3]  = (_Float16)x0.w;
    f[4]  = (_Float16)x1.x; f[5]  = (_Float16)x1.y; f[6]  = (_Float16)x1.z; f[7]  = (_Float16)x1.w;
    f[8]  = (_Float16)x2.x; f[9]  = (_Float16)x2.y; f[10] = (_Float16)x2.z; f[11] = (_Float16)x2.w;
    f[12] = (_Float16)x3.x; f[13] = (_Float16)x3.y; f[14] = (_Float16)x3.z; f[15] = (_Float16)x3.w;
    return f;
}

// Branchless guarded loader (row-major along K).  Addresses are clamped into
// the valid region so loads are UNCONDITIONAL; out-of-range elements are
// zeroed via multiply by a {0,1} mask.  The multiply consumes the loaded
// value unconditionally, so the compiler cannot sink the load into a branch.
__device__ __forceinline__ v16h load_frag_clamp_row(
    const float* __restrict__ X, int rows, int K, int ld, int r, int k0, int hs)
{
    const bool  rok   = r < rows;
    const int   rc    = rok ? r : (rows - 1);
    const float rmask = rok ? 1.f : 0.f;
    const float* base = X + (size_t)rc * ld;
    v16h f;
#pragma unroll
    for (int i = 0; i < 16; ++i) {
        const int   kk    = k0 + (i & 7) + ((i >> 3) << 4) + (hs << 3);
        const int   kc    = kk < K ? kk : (K - 1);
        const float kmask = kk < K ? rmask : 0.f;
        f[i] = (_Float16)(base[kc] * kmask);
    }
    return f;
}

// (K,N) row-major B, fixed column n: strided along K, coalesced across lanes
__device__ __forceinline__ v16h load_fragB_kn(
    const float* __restrict__ B, int ldb, int n, int k0, int hs)
{
    v16h f;
#pragma unroll
    for (int i = 0; i < 16; ++i) {
        const int kk = k0 + (i & 7) + ((i >> 3) << 4) + (hs << 3);
        f[i] = (_Float16)B[(size_t)kk * ldb + n];
    }
    return f;
}

// branchless clamped variant (mask-multiply zeroing, unconditional loads)
__device__ __forceinline__ v16h load_fragB_kn_clamp(
    const float* __restrict__ B, int N, int K, int ldb, int n, int k0, int hs)
{
    const bool  nok   = n < N;
    const int   nc    = nok ? n : (N - 1);
    const float nmask = nok ? 1.f : 0.f;
    v16h f;
#pragma unroll
    for (int i = 0; i < 16; ++i) {
        const int   kk    = k0 + (i & 7) + ((i >> 3) << 4) + (hs << 3);
        const int   kc    = kk < K ? kk : (K - 1);
        const float kmask = kk < K ? nmask : 0.f;
        f[i] = (_Float16)(B[(size_t)kc * ldb + nc] * kmask);
    }
    return f;
}

__device__ __forceinline__ void store_tile(
    float* __restrict__ C, const float* __restrict__ bias,
    const float* __restrict__ addsrc, int M, int N, int ldc,
    int tm, int tn, int lr, int hs, v8f acc)
{
    const int cn = tn + lr;
    if (cn >= N) return;
    const float bsv = bias ? bias[cn] : 0.f;
#pragma unroll
    for (int vi = 0; vi < 8; ++vi) {
        const int cm = tm + vi + (hs << 3);
        if (cm < M) {
            float v = acc[vi] + bsv;
            if (addsrc) v += addsrc[(size_t)cm * ldc + cn];
            C[(size_t)cm * ldc + cn] = v;
        }
    }
}

#define WMMA_F16(Afrag, Bfrag, Cacc) \
    __builtin_amdgcn_wmma_f32_16x16x32_f16(false, (Afrag), false, (Bfrag), (short)0, (Cacc), false, false)

// =====================================================================
// GEMM: one wave per 32x32 C tile -> 4 accumulators, 4 WMMAs per K-step.
// C[M,N] = A[M,K] @ B (+bias[n]) (+addsrc[m,n])
// TRANSB=false: B is (K,N) row-major (ldb).  TRANSB=true: B is (N,K).
// Path selection (interior vs edge) hoisted OUT of the K-loop: each inner
// loop is branch-free with 4 back-to-back WMMAs.
// =====================================================================
template <bool TRANSB>
__global__ __launch_bounds__(32) void k_gemm_wmma(
    const float* __restrict__ A, const float* __restrict__ B,
    const float* __restrict__ bias, const float* __restrict__ addsrc,
    float* __restrict__ C,
    int M, int N, int K, int lda, int ldb, int ldc)
{
    const int tn   = blockIdx.x * 32;
    const int tm   = blockIdx.y * 32;
    const int lane = threadIdx.x;
    const int hs   = lane >> 4;
    const int lr   = lane & 15;

    const int am0 = tm + lr, am1 = am0 + 16;
    const int bn0 = tn + lr, bn1 = bn0 + 16;

    const bool interior = (tm + 32 <= M) && (tn + 32 <= N) && ((K & 31) == 0) &&
                          ((lda & 3) == 0) && (!TRANSB || ((ldb & 3) == 0));

    v8f acc00 = {}, acc01 = {}, acc10 = {}, acc11 = {};

    if (interior) {
        const float* pa0 = A + (size_t)am0 * lda + (hs << 3);
        const float* pa1 = A + (size_t)am1 * lda + (hs << 3);
        if (TRANSB) {
            const float* pb0 = B + (size_t)bn0 * ldb + (hs << 3);
            const float* pb1 = B + (size_t)bn1 * ldb + (hs << 3);
            for (int k0 = 0; k0 < K; k0 += 32) {
                __builtin_prefetch(pa0 + k0 + 32, 0, 0);   // global_prefetch next slab
                v16h a0 = load_frag_contig(pa0 + k0);
                v16h a1 = load_frag_contig(pa1 + k0);
                v16h b0 = load_frag_contig(pb0 + k0);
                v16h b1 = load_frag_contig(pb1 + k0);
                acc00 = WMMA_F16(a0, b0, acc00);
                acc01 = WMMA_F16(a0, b1, acc01);
                acc10 = WMMA_F16(a1, b0, acc10);
                acc11 = WMMA_F16(a1, b1, acc11);
            }
        } else {
            for (int k0 = 0; k0 < K; k0 += 32) {
                __builtin_prefetch(pa0 + k0 + 32, 0, 0);   // global_prefetch next slab
                v16h a0 = load_frag_contig(pa0 + k0);
                v16h a1 = load_frag_contig(pa1 + k0);
                v16h b0 = load_fragB_kn(B, ldb, bn0, k0, hs);
                v16h b1 = load_fragB_kn(B, ldb, bn1, k0, hs);
                acc00 = WMMA_F16(a0, b0, acc00);
                acc01 = WMMA_F16(a0, b1, acc01);
                acc10 = WMMA_F16(a1, b0, acc10);
                acc11 = WMMA_F16(a1, b1, acc11);
            }
        }
    } else {
        for (int k0 = 0; k0 < K; k0 += 32) {
            v16h a0 = load_frag_clamp_row(A, M, K, lda, am0, k0, hs);
            v16h a1 = load_frag_clamp_row(A, M, K, lda, am1, k0, hs);
            v16h b0, b1;
            if (TRANSB) {
                b0 = load_frag_clamp_row(B, N, K, ldb, bn0, k0, hs);
                b1 = load_frag_clamp_row(B, N, K, ldb, bn1, k0, hs);
            } else {
                b0 = load_fragB_kn_clamp(B, N, K, ldb, bn0, k0, hs);
                b1 = load_fragB_kn_clamp(B, N, K, ldb, bn1, k0, hs);
            }
            acc00 = WMMA_F16(a0, b0, acc00);
            acc01 = WMMA_F16(a0, b1, acc01);
            acc10 = WMMA_F16(a1, b0, acc10);
            acc11 = WMMA_F16(a1, b1, acc11);
        }
    }

    store_tile(C, bias, addsrc, M, N, ldc, tm,      tn,      lr, hs, acc00);
    store_tile(C, bias, addsrc, M, N, ldc, tm,      tn + 16, lr, hs, acc01);
    store_tile(C, bias, addsrc, M, N, ldc, tm + 16, tn,      lr, hs, acc10);
    store_tile(C, bias, addsrc, M, N, ldc, tm + 16, tn + 16, lr, hs, acc11);
}

// ---------------- tokenizer: zero-padded reshape ----------------
__global__ void k_build_tokens(const float* __restrict__ x, float* __restrict__ out,
                               int Bn, int in_dim, int ntok, int tokdim)
{
    int idx = blockIdx.x * blockDim.x + threadIdx.x;
    int total = Bn * ntok * tokdim;
    if (idx >= total) return;
    int j = idx % tokdim;
    int t = (idx / tokdim) % ntok;
    int b = idx / (tokdim * ntok);
    int src = t * tokdim + j;
    out[idx] = (src < in_dim) ? x[(size_t)b * in_dim + src] : 0.f;
}

// ---------------- causal depthwise conv (width 4) + SiLU ----------------
__global__ void k_conv_silu(const float* __restrict__ zx, const float* __restrict__ cw,
                            const float* __restrict__ cb, float* __restrict__ out,
                            int Bn, int L)
{
    int idx = blockIdx.x * blockDim.x + threadIdx.x;
    int total = Bn * L * CONVDIM;
    if (idx >= total) return;
    int c = idx % CONVDIM;
    int l = (idx / CONVDIM) % L;
    int b = idx / (CONVDIM * L);
    float s = cb[c];
#pragma unroll
    for (int k = 0; k < DCONV; ++k) {
        int ls = l + k - (DCONV - 1);
        if (ls >= 0)
            s += cw[c * DCONV + k] * zx[(size_t)(b * L + ls) * DPROJ + DINNER + c];
    }
    out[idx] = s / (1.f + __expf(-s));
}

// ---------------- sequential SSM scan, one block per (batch, head) --------
__global__ __launch_bounds__(256) void k_ssm_scan(
    const float* __restrict__ zx, const float* __restrict__ xbc,
    const float* __restrict__ dt_bias, const float* __restrict__ A_log,
    const float* __restrict__ Dv, float* __restrict__ y, int L)
{
    const int b   = blockIdx.x >> 4;
    const int h   = blockIdx.x & 15;
    const int tid = threadIdx.x;
    const int p   = tid >> 2;
    const int nq  = tid & 3;
    __shared__ float Bsh[DSTATE];
    __shared__ float Csh[DSTATE];

    float st[32];
#pragma unroll
    for (int j = 0; j < 32; ++j) st[j] = 0.f;

    const float Ah  = -__expf(A_log[h]);
    const float dbh = dt_bias[h];
    const float Dh  = Dv[h];

    for (int t = 0; t < L; ++t) {
        const size_t rx = (size_t)(b * L + t);
        if (tid < DSTATE)      Bsh[tid]          = xbc[rx * CONVDIM + DINNER + tid];
        else                   Csh[tid - DSTATE] = xbc[rx * CONVDIM + DINNER + DSTATE + (tid - DSTATE)];
        __syncthreads();

        float dtv = zx[rx * DPROJ + DINNER + CONVDIM + h] + dbh;
        dtv = (dtv > 20.f) ? dtv : log1pf(__expf(dtv));          // softplus
        const float dA  = __expf(dtv * Ah);
        const float xv  = xbc[rx * CONVDIM + h * HEADDIM + p];
        const float dtx = dtv * xv;

        float part = 0.f;
        const int nb = nq * 32;
#pragma unroll
        for (int j = 0; j < 32; ++j) {
            st[j] = dA * st[j] + dtx * Bsh[nb + j];
            part += st[j] * Csh[nb + j];
        }
        part += __shfl_xor(part, 1);
        part += __shfl_xor(part, 2);
        if (nq == 0)
            y[rx * DINNER + h * HEADDIM + p] = part + Dh * xv;
        __syncthreads();
    }
}

// ---------------- y *= silu(z); RMS-norm * norm_w (in place on y) ---------
__global__ __launch_bounds__(256) void k_gate_rmsnorm(
    const float* __restrict__ zx, const float* __restrict__ norm_w,
    float* __restrict__ y)
{
    const int r   = blockIdx.x;
    const int tid = threadIdx.x;
    __shared__ float red[256];
    float v[4];
    float ss = 0.f;
#pragma unroll
    for (int k = 0; k < 4; ++k) {
        int c = tid + k * 256;
        float z  = zx[(size_t)r * DPROJ + c];
        float yv = y[(size_t)r * DINNER + c] * (z / (1.f + __expf(-z)));
        v[k] = yv;
        ss += yv * yv;
    }
    red[tid] = ss;
    __syncthreads();
    for (int s = 128; s > 0; s >>= 1) {
        if (tid < s) red[tid] += red[tid + s];
        __syncthreads();
    }
    const float rms = rsqrtf(red[0] / (float)DINNER + EPSF);
#pragma unroll
    for (int k = 0; k < 4; ++k) {
        int c = tid + k * 256;
        y[(size_t)r * DINNER + c] = v[k] * rms * norm_w[c];
    }
}

// ---------------- LayerNorm (general width) ----------------
__global__ __launch_bounds__(256) void k_layernorm(
    const float* __restrict__ x, const float* __restrict__ w,
    const float* __restrict__ bb, float* __restrict__ out, int W)
{
    const int r   = blockIdx.x;
    const int tid = threadIdx.x;
    __shared__ float rs[256];
    __shared__ float rq[256];
    float s = 0.f, q = 0.f;
    for (int c = tid; c < W; c += 256) {
        float xv = x[(size_t)r * W + c];
        s += xv;
        q += xv * xv;
    }
    rs[tid] = s; rq[tid] = q;
    __syncthreads();
    for (int st = 128; st > 0; st >>= 1) {
        if (tid < st) { rs[tid] += rs[tid + st]; rq[tid] += rq[tid + st]; }
        __syncthreads();
    }
    const float mean = rs[0] / (float)W;
    const float var  = rq[0] / (float)W - mean * mean;
    const float inv  = rsqrtf(var + EPSF);
    for (int c = tid; c < W; c += 256) {
        float xv = x[(size_t)r * W + c];
        out[(size_t)r * W + c] = (xv - mean) * inv * w[c] + bb[c];
    }
}

// ---------------- attention: scores / softmax / attn*V ----------------
__global__ void k_attn_scores(const float* __restrict__ qh, const float* __restrict__ kh,
                              float* __restrict__ sc, int Bn, int Lq, int Lk)
{
    int idx = blockIdx.x * blockDim.x + threadIdx.x;
    int total = Bn * CAHEADS * Lq * Lk;
    if (idx >= total) return;
    int ki = idx % Lk;
    int qi = (idx / Lk) % Lq;
    int h  = (idx / (Lk * Lq)) % CAHEADS;
    int b  = idx / (Lk * Lq * CAHEADS);
    float s = 0.f;
#pragma unroll 8
    for (int d = 0; d < HEADDIM; ++d)
        s += qh[(size_t)(b * Lq + qi) * DMODEL + h * HEADDIM + d] *
             kh[(size_t)(b * Lk + ki) * DMODEL + h * HEADDIM + d];
    sc[idx] = s * 0.125f;   // 1/sqrt(64)
}

__global__ void k_softmax(float* __restrict__ sc, int Bn, int Lq, int Lk)
{
    int idx = blockIdx.x * blockDim.x + threadIdx.x;
    int total = Bn * CAHEADS * Lq;
    if (idx >= total) return;
    float* row = sc + (size_t)idx * Lk;
    float m = -1e30f;
    for (int k = 0; k < Lk; ++k) m = fmaxf(m, row[k]);
    float sum = 0.f;
    for (int k = 0; k < Lk; ++k) { float e = __expf(row[k] - m); row[k] = e; sum += e; }
    float inv = 1.f / sum;
    for (int k = 0; k < Lk; ++k) row[k] *= inv;
}

__global__ void k_attn_v(const float* __restrict__ sc, const float* __restrict__ vh,
                         float* __restrict__ ao, int Bn, int Lq, int Lk)
{
    int idx = blockIdx.x * blockDim.x + threadIdx.x;
    int total = Bn * Lq * DMODEL;
    if (idx >= total) return;
    int c  = idx % DMODEL;
    int qi = (idx / DMODEL) % Lq;
    int b  = idx / (DMODEL * Lq);
    int h  = c >> 6;
    int d  = c & 63;
    float s = 0.f;
    for (int ki = 0; ki < Lk; ++ki)
        s += sc[(size_t)((b * CAHEADS + h) * Lq + qi) * Lk + ki] *
             vh[(size_t)(b * Lk + ki) * DMODEL + h * HEADDIM + d];
    ao[idx] = s;
}

// ---------------- mean-pool over tokens + concat ----------------
__global__ void k_meanpool(const float* __restrict__ xs, const float* __restrict__ xg,
                           float* __restrict__ fused)
{
    int idx = blockIdx.x * blockDim.x + threadIdx.x;
    if (idx >= BATCH * 2 * DMODEL) return;
    int c = idx % (2 * DMODEL);
    int b = idx / (2 * DMODEL);
    float s = 0.f;
    if (c < DMODEL) {
        for (int l = 0; l < NTOK_S; ++l) s += xs[(size_t)(b * NTOK_S + l) * DMODEL + c];
        s /= (float)NTOK_S;
    } else {
        int cc = c - DMODEL;
        for (int l = 0; l < NTOK_G; ++l) s += xg[(size_t)(b * NTOK_G + l) * DMODEL + cc];
        s /= (float)NTOK_G;
    }
    fused[idx] = s;
}

// =====================================================================
// host-side orchestration
// =====================================================================
static inline void gemm(hipStream_t st, const float* A, const float* B,
                        const float* bias, const float* addsrc, float* C,
                        int M, int N, int K, int lda, int ldb, int ldc, int transB)
{
    dim3 g((N + 31) / 32, (M + 31) / 32);
    if (transB)
        k_gemm_wmma<true><<<g, dim3(32), 0, st>>>(A, B, bias, addsrc, C, M, N, K, lda, ldb, ldc);
    else
        k_gemm_wmma<false><<<g, dim3(32), 0, st>>>(A, B, bias, addsrc, C, M, N, K, lda, ldb, ldc);
}

static void mamba_layer(hipStream_t st, const float* x, float* xnext, int Bn, int L,
                        const float* Win, const float* cw, const float* cb,
                        const float* dtb, const float* Alog, const float* Dv,
                        const float* nw, const float* Wout,
                        float* zx, float* xbc, float* y)
{
    const int rows = Bn * L;
    gemm(st, x, Win, nullptr, nullptr, zx, rows, DPROJ, DMODEL, DMODEL, DPROJ, DPROJ, 0);
    const int tconv = rows * CONVDIM;
    k_conv_silu<<<(tconv + 255) / 256, 256, 0, st>>>(zx, cw, cb, xbc, Bn, L);
    k_ssm_scan<<<Bn * NHEADS, 256, 0, st>>>(zx, xbc, dtb, Alog, Dv, y, L);
    k_gate_rmsnorm<<<rows, 256, 0, st>>>(zx, nw, y);
    gemm(st, y, Wout, nullptr, nullptr, xnext, rows, DMODEL, DINNER, DINNER, DMODEL, DMODEL, 0);
}

static void cross_attn(hipStream_t st, const float* q, const float* kv,
                       int Bn, int Lq, int Lk,
                       const float* lnw, const float* lnb,
                       const float* inw, const float* inb,
                       const float* outw, const float* outb,
                       float* qln, float* qh, float* kh, float* vh,
                       float* sc, float* ao, float* qout)
{
    k_layernorm<<<Bn * Lq, 256, 0, st>>>(q, lnw, lnb, qln, DMODEL);
    // q/k/v projections: y = x @ W^T + b
    gemm(st, qln, inw,                       inb,              nullptr, qh, Bn * Lq, DMODEL, DMODEL, DMODEL, DMODEL, DMODEL, 1);
    gemm(st, kv,  inw + DMODEL * DMODEL,     inb + DMODEL,     nullptr, kh, Bn * Lk, DMODEL, DMODEL, DMODEL, DMODEL, DMODEL, 1);
    gemm(st, kv,  inw + 2 * DMODEL * DMODEL, inb + 2 * DMODEL, nullptr, vh, Bn * Lk, DMODEL, DMODEL, DMODEL, DMODEL, DMODEL, 1);
    const int tsc = Bn * CAHEADS * Lq * Lk;
    k_attn_scores<<<(tsc + 255) / 256, 256, 0, st>>>(qh, kh, sc, Bn, Lq, Lk);
    const int trow = Bn * CAHEADS * Lq;
    k_softmax<<<(trow + 255) / 256, 256, 0, st>>>(sc, Bn, Lq, Lk);
    const int tav = Bn * Lq * DMODEL;
    k_attn_v<<<(tav + 255) / 256, 256, 0, st>>>(sc, vh, ao, Bn, Lq, Lk);
    // out projection + residual
    gemm(st, ao, outw, outb, q, qout, Bn * Lq, DMODEL, DMODEL, DMODEL, DMODEL, DMODEL, 1);
}

extern "C" void kernel_launch(void* const* d_in, const int* in_sizes, int n_in,
                              void* d_out, int out_size, void* d_ws, size_t ws_size,
                              hipStream_t stream)
{
    (void)in_sizes; (void)n_in; (void)out_size; (void)ws_size;
    // -------- inputs (setup_inputs order) --------
    const float* x_spectra = (const float*)d_in[0];
    const float* x_gaia    = (const float*)d_in[1];
    const float* tok_w_s   = (const float*)d_in[2];
    const float* tok_b_s   = (const float*)d_in[3];
    const float* tok_w_g   = (const float*)d_in[4];
    const float* tok_b_g   = (const float*)d_in[5];
    const float* ms[8];  for (int i = 0; i < 8; ++i) ms[i] = (const float*)d_in[6 + i];
    const float* mg[8];  for (int i = 0; i < 8; ++i) mg[i] = (const float*)d_in[14 + i];
    const float* cas[6]; for (int i = 0; i < 6; ++i) cas[i] = (const float*)d_in[22 + i];
    const float* cag[6]; for (int i = 0; i < 6; ++i) cag[i] = (const float*)d_in[28 + i];
    const float* cls_ln_w = (const float*)d_in[34];
    const float* cls_ln_b = (const float*)d_in[35];
    const float* cls_w    = (const float*)d_in[36];
    const float* cls_b    = (const float*)d_in[37];
    float* out = (float*)d_out;

    // -------- workspace bump allocator --------
    float* ws = (float*)d_ws;
    size_t off = 0;
    auto alloc = [&](size_t n) { float* p = ws + off; off += n; return p; };
    const int RS = BATCH * NTOK_S;   // 456 spectra token rows
    const int RG = BATCH * NTOK_G;   // 72 gaia token rows
    float* Xtok = alloc((size_t)RS * TOK_S);
    float* xsA  = alloc((size_t)RS * DMODEL);
    float* xsB  = alloc((size_t)RS * DMODEL);
    float* xgA  = alloc((size_t)RG * DMODEL);
    float* xgB  = alloc((size_t)RG * DMODEL);
    float* zx   = alloc((size_t)RS * DPROJ);
    float* xbc  = alloc((size_t)RS * CONVDIM);
    float* ybuf = alloc((size_t)RS * DINNER);
    float* qln  = alloc((size_t)RS * DMODEL);
    float* qh   = alloc((size_t)RS * DMODEL);
    float* kh   = alloc((size_t)RS * DMODEL);
    float* vh   = alloc((size_t)RS * DMODEL);
    float* sc   = alloc((size_t)BATCH * CAHEADS * NTOK_S * NTOK_S);
    float* ao   = alloc((size_t)RS * DMODEL);
    float* fus  = alloc((size_t)BATCH * 2 * DMODEL);
    float* fln  = alloc((size_t)BATCH * 2 * DMODEL);

    // -------- tokenize --------
    const int ttok = RS * TOK_S;
    k_build_tokens<<<(ttok + 255) / 256, 256, 0, stream>>>(x_spectra, Xtok, BATCH, IN_S, NTOK_S, TOK_S);
    gemm(stream, Xtok, tok_w_s, tok_b_s, nullptr, xsA, RS, DMODEL, TOK_S, TOK_S, DMODEL, DMODEL, 0);
    // gaia: 18 = 9*2 exactly -> reshape is a no-op, use x_gaia directly
    gemm(stream, x_gaia, tok_w_g, tok_b_g, nullptr, xgA, RG, DMODEL, TOK_G, TOK_G, DMODEL, DMODEL, 0);

    // -------- mamba stacks (ping-pong buffers; 10 layers -> ends in *A) ----
    {
        float* cur = xsA; float* nxt = xsB;
        for (int i = 0; i < NLAYERS; ++i) {
            mamba_layer(stream, cur, nxt, BATCH, NTOK_S,
                        ms[0] + (size_t)i * DMODEL * DPROJ,
                        ms[1] + (size_t)i * CONVDIM * DCONV,
                        ms[2] + (size_t)i * CONVDIM,
                        ms[3] + (size_t)i * NHEADS,
                        ms[4] + (size_t)i * NHEADS,
                        ms[5] + (size_t)i * NHEADS,
                        ms[6] + (size_t)i * DINNER,
                        ms[7] + (size_t)i * DINNER * DMODEL,
                        zx, xbc, ybuf);
            float* t = cur; cur = nxt; nxt = t;
        }
    }
    {
        float* cur = xgA; float* nxt = xgB;
        for (int i = 0; i < NLAYERS; ++i) {
            mamba_layer(stream, cur, nxt, BATCH, NTOK_G,
                        mg[0] + (size_t)i * DMODEL * DPROJ,
                        mg[1] + (size_t)i * CONVDIM * DCONV,
                        mg[2] + (size_t)i * CONVDIM,
                        mg[3] + (size_t)i * NHEADS,
                        mg[4] + (size_t)i * NHEADS,
                        mg[5] + (size_t)i * NHEADS,
                        mg[6] + (size_t)i * DINNER,
                        mg[7] + (size_t)i * DINNER * DMODEL,
                        zx, xbc, ybuf);
            float* t = cur; cur = nxt; nxt = t;
        }
    }

    // -------- cross attention (xs first, then xg attends to updated xs) ----
    cross_attn(stream, xsA, xgA, BATCH, NTOK_S, NTOK_G,
               cas[0], cas[1], cas[2], cas[3], cas[4], cas[5],
               qln, qh, kh, vh, sc, ao, xsB);
    cross_attn(stream, xgA, xsB, BATCH, NTOK_G, NTOK_S,
               cag[0], cag[1], cag[2], cag[3], cag[4], cag[5],
               qln, qh, kh, vh, sc, ao, xgB);

    // -------- pool, layernorm, classifier --------
    const int tmp = BATCH * 2 * DMODEL;
    k_meanpool<<<(tmp + 255) / 256, 256, 0, stream>>>(xsB, xgB, fus);
    k_layernorm<<<BATCH, 256, 0, stream>>>(fus, cls_ln_w, cls_ln_b, fln, 2 * DMODEL);
    // logits = fln @ cls_w^T + cls_b   (cls_w is (55, 1024))
    gemm(stream, fln, cls_w, cls_b, nullptr, out, BATCH, NCLS, 2 * DMODEL,
         2 * DMODEL, 2 * DMODEL, NCLS, 1);
}
